// AttentionEncoderLayer_59236188946622
// MI455X (gfx1250) — matrix operand order repaired and checked by reading, not compiled, not run on hardware
//
#include <hip/hip_runtime.h>
#include <hip/hip_bf16.h>

// ---------------------------------------------------------------------------
// Types for WMMA fragments (wave32, gfx1250)
// ---------------------------------------------------------------------------
typedef _Float16 h16v __attribute__((ext_vector_type(16)));
typedef _Float16 h8v  __attribute__((ext_vector_type(8)));
typedef float    f8v  __attribute__((ext_vector_type(8)));
typedef float    f4v  __attribute__((ext_vector_type(4)));

#define WMMA_F32_F16(a, b, c) \
    __builtin_amdgcn_wmma_f32_16x16x32_f16(false, (a), false, (b), (short)0, (c), false, false)

static __device__ __forceinline__ f8v zero8() {
    f8v z = {0.f, 0.f, 0.f, 0.f, 0.f, 0.f, 0.f, 0.f};
    return z;
}

// Compiler-level reordering fence. Hardware keeps same-wave DS ops in order
// (per the CDNA5 ISA) and the backend inserts
// s_wait_dscnt for VGPR consumers, so wave-private LDS RAW/WAR needs no
// block barrier -- only a scheduling fence.
static __device__ __forceinline__ void wave_lds_fence() {
    asm volatile("" ::: "memory");
    __builtin_amdgcn_wave_barrier();
}

// Problem constants
#define BB   4
#define SS   2048
#define HID  1024
#define NH   16
#define HD   64
#define MTOT (BB * SS)   // 8192

// ---------------------------------------------------------------------------
// Kernel 1: projection GEMM   out[m][n] = H[m][:] . W[n][:] + bias[n]
// H: [8192,1024] f32 row-major, W: [1024,1024] f32 row-major (so W^T GEMM has
// contiguous B-fragment loads). Each wave computes a 16x64 tile with 4 f32
// accumulators via v_wmma_f32_16x16x32_f16.
// transposed==0 : store f16 at out[m*1024 + n]              ([b,s,h,d] layout)
// transposed==1 : store f16 at out[((b*16+h)*64+d)*2048+s]  (V^T layout)
// ---------------------------------------------------------------------------
__global__ __launch_bounds__(128) void proj_gemm_kernel(
    const float* __restrict__ H, const float* __restrict__ W,
    const float* __restrict__ bias, _Float16* __restrict__ out, int transposed)
{
    const int lane = threadIdx.x & 31;
    const int wave = threadIdx.x >> 5;
    const int lo   = lane & 15;
    const bool hi  = lane >= 16;

    const int m0     = blockIdx.y * 16;
    const int n_base = (blockIdx.x * 4 + wave) * 64;

    const float* Hrow = H + (size_t)(m0 + lo) * HID;

    f8v acc[4] = {zero8(), zero8(), zero8(), zero8()};

    for (int k0 = 0; k0 < HID; k0 += 32) {
        // ---- A fragment (16x32 f16): lane lo = row; per-lane K halves
        const int ka = k0 + (hi ? 16 : 0);
        f4v a0 = *(const f4v*)(Hrow + ka);
        f4v a1 = *(const f4v*)(Hrow + ka + 4);
        f4v a2 = *(const f4v*)(Hrow + ka + 8);
        f4v a3 = *(const f4v*)(Hrow + ka + 12);
        h16v afrag;
#pragma unroll
        for (int e = 0; e < 4; ++e) {
            afrag[e]      = (_Float16)a0[e];
            afrag[e + 4]  = (_Float16)a1[e];
            afrag[e + 8]  = (_Float16)a2[e];
            afrag[e + 12] = (_Float16)a3[e];
        }
        // ---- 4 B fragments (32x16 f16), contiguous in W[n][k]
#pragma unroll
        for (int j = 0; j < 4; ++j) {
            const float* Wrow = W + (size_t)(n_base + 16 * j + lo) * HID + k0 + (hi ? 16 : 0);
            f4v b0 = *(const f4v*)(Wrow);
            f4v b1 = *(const f4v*)(Wrow + 4);
            f4v b2 = *(const f4v*)(Wrow + 8);
            f4v b3 = *(const f4v*)(Wrow + 12);
            h16v bfrag;
#pragma unroll
            for (int e = 0; e < 4; ++e) {
                bfrag[e]      = (_Float16)b0[e];
                bfrag[e + 4]  = (_Float16)b1[e];
                bfrag[e + 8]  = (_Float16)b2[e];
                bfrag[e + 12] = (_Float16)b3[e];
            }
            acc[j] = WMMA_F32_F16(afrag, bfrag, acc[j]);
        }
    }

    // ---- bias + store. D layout: lane column n = 16j+lo, VGPR i row = i + (hi?8:0)
#pragma unroll
    for (int j = 0; j < 4; ++j) {
        const int n  = n_base + 16 * j + lo;
        const float bv = bias[n];
        if (!transposed) {
#pragma unroll
            for (int i = 0; i < 8; ++i) {
                const int row = m0 + i + (hi ? 8 : 0);
                out[(size_t)row * HID + n] = (_Float16)(acc[j][i] + bv);
            }
        } else {
            // V^T: [b][h][d][s], 8 consecutive s per lane -> one 16B store
            const int b = m0 >> 11;                    // m0 / SS
            const int s = (m0 & (SS - 1)) + (hi ? 8 : 0);
            const int h = n >> 6, d = n & 63;
            h8v v;
#pragma unroll
            for (int i = 0; i < 8; ++i) v[i] = (_Float16)(acc[j][i] + bv);
            *(h8v*)(out + ((size_t)((b * NH + h) * HD + d) * SS + s)) = v;
        }
    }
}

// ---------------------------------------------------------------------------
// Kernel 2: kmod = (k/8 + softsign(softsign(k)/8) + v) / 8, in place over Kh.
// (the trailing /8 folds the reference's final score scale into K.)
// ---------------------------------------------------------------------------
__global__ __launch_bounds__(256) void kmod_kernel(
    _Float16* __restrict__ Kh, const _Float16* __restrict__ Vt)
{
    const unsigned idx = blockIdx.x * 256u + threadIdx.x;  // 0 .. 8388607
    const int n = idx & (HID - 1);
    const unsigned m = idx >> 10;
    const int b = (int)(m >> 11);
    const int s = (int)(m & (SS - 1));
    const int h = n >> 6, d = n & 63;

    const float k = (float)Kh[idx];
    const float v = (float)Vt[(size_t)((b * NH + h) * HD + d) * SS + s];

    const float ss1 = k / (1.0f + fabsf(k));
    const float t   = ss1 * 0.125f;
    const float ss2 = t / (1.0f + fabsf(t));
    const float km  = k * 0.125f + ss2 + v;
    Kh[idx] = (_Float16)(km * 0.125f);
}

// ---------------------------------------------------------------------------
// Kernel 3: flash attention. Block = 8 independent waves; each wave owns 16
// query rows and streams 2048 keys in tiles of 32 with online softmax:
//   4 WMMA for scores (2 key subtiles x 2 d-chunks), 4 WMMA for O += P.V
// P is moved D-layout -> A-layout through a wave-private LDS tile (16x40 f16,
// padded so the 16-row strided reads hit distinct banks); ordering relies on
// the in-order same-wave DS guarantee (no block barriers).
// Query-dim masking per reference (mask[b,q]==0 -> whole score row = -1e4).
// ---------------------------------------------------------------------------
__global__ __launch_bounds__(256) void attn_kernel(
    const _Float16* __restrict__ Qh, const _Float16* __restrict__ Km,
    const _Float16* __restrict__ Vt, const int* __restrict__ mask,
    float* __restrict__ out)
{
    __shared__ __attribute__((aligned(16))) _Float16 plds[8][16][40];

    const int lane = threadIdx.x & 31;
    const int wave = threadIdx.x >> 5;
    const int lo   = lane & 15;
    const bool hi  = lane >= 16;

    const int bh = blockIdx.x;            // 0..63
    const int b  = bh >> 4, h = bh & 15;
    const int q0 = blockIdx.y * 128 + wave * 16;

    // ---- Q A-fragments for d-chunks [0,32) and [32,64)
    const _Float16* Qrow = Qh + (size_t)(b * SS + q0 + lo) * HID + h * HD;
    const int ka = hi ? 16 : 0;
    h8v ql0 = *(const h8v*)(Qrow + ka);
    h8v qh0 = *(const h8v*)(Qrow + 8 + ka);
    h8v ql1 = *(const h8v*)(Qrow + 32 + ka);
    h8v qh1 = *(const h8v*)(Qrow + 40 + ka);
    h16v aq0, aq1;
#pragma unroll
    for (int e = 0; e < 8; ++e) {
        aq0[e] = ql0[e]; aq0[e + 8] = qh0[e];
        aq1[e] = ql1[e]; aq1[e + 8] = qh1[e];
    }

    // ---- query-dim mask per owned row (row = i + (hi?8:0))
    bool qm[8];
#pragma unroll
    for (int i = 0; i < 8; ++i)
        qm[i] = (mask[b * SS + q0 + i + (hi ? 8 : 0)] != 0);

    f8v o[4] = {zero8(), zero8(), zero8(), zero8()};
    float mrow[8], lrow[8];
#pragma unroll
    for (int i = 0; i < 8; ++i) { mrow[i] = -1e30f; lrow[i] = 0.0f; }

    const _Float16* Kbase = Km + (size_t)(b * SS) * HID + h * HD;
    const _Float16* Vbase = Vt + (size_t)bh * HD * SS;

    for (int k0 = 0; k0 < SS; k0 += 32) {
        // ---- scores: two 16x16 tiles (keys k0..+15, k0+16..+31)
        const _Float16* kr1 = Kbase + (size_t)(k0 + lo) * HID + (hi ? 16 : 0);
        const _Float16* kr2 = Kbase + (size_t)(k0 + 16 + lo) * HID + (hi ? 16 : 0);
        h16v bk0 = *(const h16v*)(kr1);
        h16v bk1 = *(const h16v*)(kr1 + 32);
        h16v bk2 = *(const h16v*)(kr2);
        h16v bk3 = *(const h16v*)(kr2 + 32);

        f8v s1 = WMMA_F32_F16(aq0, bk0, zero8());
        s1     = WMMA_F32_F16(aq1, bk1, s1);
        f8v s2 = WMMA_F32_F16(aq0, bk2, zero8());
        s2     = WMMA_F32_F16(aq1, bk3, s2);

        // ---- online softmax (row = i + (hi?8:0); its 16 cols live across the
        //      16-lane half-group, so butterfly with masks 1,2,4,8 stays in-row)
#pragma unroll
        for (int i = 0; i < 8; ++i) {
            float a = qm[i] ? s1[i] : -10000.0f;
            float c = qm[i] ? s2[i] : -10000.0f;
            float r = fmaxf(a, c);
#pragma unroll
            for (int off = 1; off < 16; off <<= 1)
                r = fmaxf(r, __shfl_xor(r, off, 32));
            const float mnew  = fmaxf(mrow[i], r);
            const float alpha = __expf(mrow[i] - mnew);
            mrow[i] = mnew;
            a = __expf(a - mnew);
            c = __expf(c - mnew);
            s1[i] = a;
            s2[i] = c;
            float ps = a + c;
#pragma unroll
            for (int off = 1; off < 16; off <<= 1)
                ps += __shfl_xor(ps, off, 32);
            lrow[i] = lrow[i] * alpha + ps;
#pragma unroll
            for (int j = 0; j < 4; ++j) o[j][i] *= alpha;
        }

        // ---- transpose P: D-layout -> wave-private LDS [row][kk]
        {
            const int r = hi ? 8 : 0;
#pragma unroll
            for (int i = 0; i < 8; ++i) {
                plds[wave][i + r][lo]      = (_Float16)s1[i];
                plds[wave][i + r][16 + lo] = (_Float16)s2[i];
            }
        }
        wave_lds_fence();   // same-wave DS ops are in order; just stop reordering

        // ---- read P as A-fragment (16x32)
        const _Float16* pr = &plds[wave][lo][0];
        const int kb = hi ? 16 : 0;
        h8v p0 = *(const h8v*)(pr + kb);
        h8v p1 = *(const h8v*)(pr + 8 + kb);
        h16v ap;
#pragma unroll
        for (int e = 0; e < 8; ++e) { ap[e] = p0[e]; ap[e + 8] = p1[e]; }
        wave_lds_fence();   // protect WAR vs next iteration's stores

        // ---- O += P . V ; V^T layout makes B-fragments contiguous 32B loads
#pragma unroll
        for (int j = 0; j < 4; ++j) {
            const _Float16* vr = Vbase + (size_t)(16 * j + lo) * SS + k0 + (hi ? 16 : 0);
            h16v bv = *(const h16v*)vr;
            o[j] = WMMA_F32_F16(ap, bv, o[j]);
        }
    }

    // ---- normalize and store: out[b, q, h*64 + d] fp32, non-temporal
    //      (d_out is write-only; keep L2 for the Q/K/V working set)
#pragma unroll
    for (int j = 0; j < 4; ++j) {
        const int col = h * HD + 16 * j + lo;
#pragma unroll
        for (int i = 0; i < 8; ++i) {
            const int qrow = q0 + i + (hi ? 8 : 0);
            __builtin_nontemporal_store(o[j][i] / lrow[i],
                                        out + (size_t)(b * SS + qrow) * HID + col);
        }
    }
}

// ---------------------------------------------------------------------------
// Launch
// ---------------------------------------------------------------------------
extern "C" void kernel_launch(void* const* d_in, const int* in_sizes, int n_in,
                              void* d_out, int out_size, void* d_ws, size_t ws_size,
                              hipStream_t stream) {
    const float* H   = (const float*)d_in[0];
    const float* Wq  = (const float*)d_in[1];
    const float* bq  = (const float*)d_in[2];
    const float* Wk  = (const float*)d_in[3];
    const float* bk  = (const float*)d_in[4];
    const float* Wv  = (const float*)d_in[5];
    const float* bv  = (const float*)d_in[6];
    const int*   am  = (const int*)d_in[7];
    float* out = (float*)d_out;

    const size_t elems = (size_t)MTOT * HID;   // 8M f16 each
    _Float16* Qh = (_Float16*)d_ws;
    _Float16* Kh = Qh + elems;
    _Float16* Vt = Kh + elems;

    dim3 pgrid(4, MTOT / 16, 1);               // (n-tiles/4, m-tiles)
    proj_gemm_kernel<<<pgrid, 128, 0, stream>>>(H, Wq, bq, Qh, 0);
    proj_gemm_kernel<<<pgrid, 128, 0, stream>>>(H, Wk, bk, Kh, 0);
    proj_gemm_kernel<<<pgrid, 128, 0, stream>>>(H, Wv, bv, Vt, 1);

    kmod_kernel<<<(unsigned)(elems / 256), 256, 0, stream>>>(Kh, Vt);

    dim3 agrid(BB * NH, SS / 128, 1);          // (64 heads, 16 q-blocks)
    attn_kernel<<<agrid, 256, 0, stream>>>(Qh, Kh, Vt, am, out);
}
